// MultiHeadSelfAttention_82016695484983
// MI455X (gfx1250) — matrix operand order repaired and checked by reading, not compiled, not run on hardware
//
#include <hip/hip_runtime.h>
#include <stdint.h>
#include <math.h>

// ---------------------------------------------------------------------------
// MultiHeadSelfAttention forward for MI455X (gfx1250, wave32, WMMA).
//   Phase 1: fused QKV projection (fp32 in -> bf16 QKV in workspace)
//            128x128 tiles, v_wmma_f32_16x16x32_bf16, LDS-staged operands.
//            V is written TRANSPOSED [B,NH,HD,S] so phase 2 can async-copy it.
//   Phase 2: flash-style attention, online softmax, S^T = K*Q^T (softmax rows
//            on the lane axis), ctx^T = V^T * P^T. Key/value chunks streamed
//            with GLOBAL_LOAD_ASYNC_TO_LDS_B128 into double-buffered LDS,
//            drained with s_wait_asynccnt. 32 WMMAs / wave / 128-key chunk.
// Workspace: 3 x 16 MB bf16.
// ---------------------------------------------------------------------------

#define HDIM 1024
#define SEQ  2048
#define NB   4
#define NHD  16
#define HD   64

typedef __attribute__((ext_vector_type(16))) __bf16 v16bf;
typedef __attribute__((ext_vector_type(8)))  float  v8f;
typedef __attribute__((ext_vector_type(4)))  unsigned int u32x4;

union Frag {            // 8 VGPRs = 16 bf16
    v16bf  v;
    u32x4  q[2];
    unsigned int u[8];
};

static __device__ __forceinline__ unsigned short f2bf(float x) {
    unsigned u = __float_as_uint(x);
    unsigned r = 0x7fffu + ((u >> 16) & 1u);   // round-to-nearest-even
    return (unsigned short)((u + r) >> 16);
}
static __device__ __forceinline__ unsigned int pk2bf(float lo, float hi) {
    return (unsigned)f2bf(lo) | ((unsigned)f2bf(hi) << 16);
}

static __device__ __forceinline__ v8f wmma_bf16(const Frag& a, const Frag& b, v8f c) {
    return __builtin_amdgcn_wmma_f32_16x16x32_bf16(false, a.v, false, b.v,
                                                   (short)0, c, false, false);
}

// per-lane 16B global -> LDS async copy (ASYNCcnt)
static __device__ __forceinline__ void async_b128(unsigned int lds_off,
                                                  unsigned long long gaddr) {
    asm volatile("global_load_async_to_lds_b128 %0, %1, off"
                 :: "v"(lds_off), "v"(gaddr) : "memory");
}
static __device__ __forceinline__ void wait_async0() {
    asm volatile("s_wait_asynccnt 0x0" ::: "memory");
}

// ---------------------------------------------------------------------------
// Phase 1: QKV projection.  grid = (M/128=64, N/128=8, 3), block = 256.
// ---------------------------------------------------------------------------
__global__ __launch_bounds__(256)
void mhsa_qkv_kernel(const float* __restrict__ X,
                     const float* __restrict__ Wq, const float* __restrict__ bq,
                     const float* __restrict__ Wk, const float* __restrict__ bk,
                     const float* __restrict__ Wv, const float* __restrict__ bv,
                     unsigned short* __restrict__ wsQ,
                     unsigned short* __restrict__ wsK,
                     unsigned short* __restrict__ wsV)
{
    __shared__ unsigned short ldsA[128 * 40];
    __shared__ unsigned short ldsB[128 * 40];

    const int mat = blockIdx.z;
    const float* W  = (mat == 0) ? Wq : (mat == 1) ? Wk : Wv;
    const float* bs = (mat == 0) ? bq : (mat == 1) ? bk : bv;
    unsigned short* ws = (mat == 0) ? wsQ : (mat == 1) ? wsK : wsV;

    const int tid  = threadIdx.x;
    const int lane = tid & 31;
    const int wv   = tid >> 5;
    const int wm   = wv >> 1;
    const int wn   = wv & 1;
    const int l16  = lane & 15;
    const int h16  = lane >> 4;

    const int rowBase = blockIdx.x * 128;
    const int colBase = blockIdx.y * 128;

    v8f acc[2][4];
#pragma unroll
    for (int i = 0; i < 2; ++i)
#pragma unroll
        for (int j = 0; j < 4; ++j)
#pragma unroll
            for (int e = 0; e < 8; ++e) acc[i][j][e] = 0.0f;

    for (int k0 = 0; k0 < HDIM; k0 += 32) {
        __syncthreads();
        { // stage A: X[rowBase..+128][k0..+32] fp32 -> bf16 LDS (row-major)
            const int r = tid >> 1, half = tid & 1;
            const float* src = X + (size_t)(rowBase + r) * HDIM + k0 + half * 16;
            unsigned int tmp[8];
#pragma unroll
            for (int i = 0; i < 4; ++i) {
                float4 f = ((const float4*)src)[i];
                tmp[2 * i]     = pk2bf(f.x, f.y);
                tmp[2 * i + 1] = pk2bf(f.z, f.w);
            }
            u32x4* dst = (u32x4*)(ldsA + r * 40 + half * 16);
            dst[0] = (u32x4){tmp[0], tmp[1], tmp[2], tmp[3]};
            dst[1] = (u32x4){tmp[4], tmp[5], tmp[6], tmp[7]};
        }
        { // stage B: W[k0..+32][colBase..+128] -> LDS transposed [n][k]
            const int k  = tid >> 3;
            const int ng = (tid & 7) * 16;
            const float* src = W + (size_t)(k0 + k) * HDIM + colBase + ng;
#pragma unroll
            for (int i = 0; i < 4; ++i) {
                float4 f = ((const float4*)src)[i];
                ldsB[(ng + 4 * i + 0) * 40 + k] = f2bf(f.x);
                ldsB[(ng + 4 * i + 1) * 40 + k] = f2bf(f.y);
                ldsB[(ng + 4 * i + 2) * 40 + k] = f2bf(f.z);
                ldsB[(ng + 4 * i + 3) * 40 + k] = f2bf(f.w);
            }
        }
        __syncthreads();

        Frag af[2], bfg[4];
#pragma unroll
        for (int mt = 0; mt < 2; ++mt) {
            const int r = wm * 32 + mt * 16 + l16;
            af[mt].q[0] = *(const u32x4*)(ldsA + r * 40 + 8 * h16);
            af[mt].q[1] = *(const u32x4*)(ldsA + r * 40 + 16 + 8 * h16);
        }
#pragma unroll
        for (int nt = 0; nt < 4; ++nt) {
            const int n = wn * 64 + nt * 16 + l16;
            bfg[nt].q[0] = *(const u32x4*)(ldsB + n * 40 + 8 * h16);
            bfg[nt].q[1] = *(const u32x4*)(ldsB + n * 40 + 16 + 8 * h16);
        }
#pragma unroll
        for (int mt = 0; mt < 2; ++mt)
#pragma unroll
            for (int nt = 0; nt < 4; ++nt)
                acc[mt][nt] = wmma_bf16(af[mt], bfg[nt], acc[mt][nt]);
    }

    // epilogue: +bias, convert; Q/K -> [B,NH,S,HD], V -> [B,NH,HD,S]
#pragma unroll
    for (int mt = 0; mt < 2; ++mt) {
#pragma unroll
        for (int nt = 0; nt < 4; ++nt) {
            const int c  = colBase + wn * 64 + nt * 16 + l16;
            const float bval = bs[c];
            const int hIdx = c >> 6, d = c & 63;
#pragma unroll
            for (int j = 0; j < 8; ++j) {
                const int m = rowBase + wm * 32 + mt * 16 + j + 8 * h16;
                const int b = m >> 11, s = m & 2047;
                const size_t idx = (mat == 2)
                    ? ((size_t)(b * NHD + hIdx) * HD + d) * SEQ + s
                    : ((size_t)(b * NHD + hIdx) * SEQ + s) * HD + d;
                ws[idx] = f2bf(acc[mt][nt][j] + bval);
            }
        }
    }
}

// ---------------------------------------------------------------------------
// Phase 2: attention.  grid = (S/128=16, NH=16, B=4), block = 256 (8 waves).
// Double-buffered LDS per chunk: K [128][72] bf16 (18432 B) @ +0,
// V^T [64][136] bf16 (17408 B) @ +18432, bias float[128] @ +35840.
// ---------------------------------------------------------------------------
#define BUFB  36352
#define SMEMB (2 * BUFB)

__global__ __launch_bounds__(256)
void mhsa_attn_kernel(const unsigned short* __restrict__ Qws,
                      const unsigned short* __restrict__ Kws,
                      const unsigned short* __restrict__ Vws,
                      const long long* __restrict__ mask,
                      float* __restrict__ out)
{
    __shared__ __align__(16) unsigned char smem[SMEMB];

    const int tid  = threadIdx.x;
    const int lane = tid & 31;
    const int wv   = tid >> 5;
    const int l16  = lane & 15;
    const int h16  = lane >> 4;

    const int qblk = blockIdx.x, h = blockIdx.y, b = blockIdx.z;
    const unsigned short* Qh = Qws + (size_t)(b * NHD + h) * SEQ * HD;
    const unsigned short* Kh = Kws + (size_t)(b * NHD + h) * SEQ * HD;
    const unsigned short* Vt = Vws + (size_t)(b * NHD + h) * HD * SEQ; // [HD][SEQ]

    const unsigned int ldsBase = (unsigned int)(uintptr_t)smem; // LDS offset

    // issue async staging of one 128-key chunk into buffer bufIdx
    auto stage = [&](int keyBase, int bufIdx) {
        const unsigned int base = ldsBase + (unsigned int)bufIdx * BUFB;
#pragma unroll
        for (int i = 0; i < 8; ++i) {
            const int n = i * 256 + tid;          // 0..2047 16B segments
            unsigned long long g;
            unsigned int l;
            if (n < 1024) {                       // K rows: 128 x 8 segs
                const int row = n >> 3, seg = n & 7;
                g = (unsigned long long)(Kh + (size_t)(keyBase + row) * HD + seg * 8);
                l = base + row * 144 + seg * 16;
            } else {                              // V^T rows: 64 x 16 segs
                const int m = n - 1024;
                const int row = m >> 4, seg = m & 15;
                g = (unsigned long long)(Vt + (size_t)row * SEQ + keyBase + seg * 8);
                l = base + 18432 + row * 272 + seg * 16;
            }
            async_b128(l, g);
        }
        if (tid < 128) {                          // mask -> additive bias
            const long long mv = mask[(size_t)b * SEQ + keyBase + tid];
            *(float*)(smem + bufIdx * BUFB + 35840 + tid * 4) =
                (1.0f - (float)mv) * -3.402823466e38f;
        }
    };

    // Q^T B-operand fragments (lane = q, VGPR pairs = contiguous d)
    Frag qf[2];
    {
        const int qrow = qblk * 128 + wv * 16 + l16;
#pragma unroll
        for (int ks = 0; ks < 2; ++ks) {
            const unsigned short* p = Qh + (size_t)qrow * HD + ks * 32 + h16 * 16;
            qf[ks].q[0] = ((const u32x4*)p)[0];
            qf[ks].q[1] = ((const u32x4*)p)[1];
        }
    }

    float mrun = -INFINITY, lrun = 0.0f;
    v8f ctx[4];
#pragma unroll
    for (int i = 0; i < 4; ++i)
#pragma unroll
        for (int e = 0; e < 8; ++e) ctx[i][e] = 0.0f;

    stage(0, 0);                                  // prologue: chunk 0

    for (int kc = 0; kc < 16; ++kc) {
        const int cur = kc & 1;
        wait_async0();                            // drain this wave's asyncs
        __syncthreads();                          // chunk kc visible to all
        if (kc < 15) stage((kc + 1) * 128, 1 - cur); // overlap next chunk

        const unsigned short* ldsK  = (const unsigned short*)(smem + cur * BUFB);
        const unsigned short* ldsVT = (const unsigned short*)(smem + cur * BUFB + 18432);
        const float*          ldsBias = (const float*)(smem + cur * BUFB + 35840);

        // ---- S^T = K * Q^T : 8 key-tiles x 2 k-steps
        v8f sc[8];
#pragma unroll
        for (int t = 0; t < 8; ++t) {
            v8f a;
#pragma unroll
            for (int e = 0; e < 8; ++e) a[e] = 0.0f;
            const int kr = t * 16 + l16;
#pragma unroll
            for (int ks = 0; ks < 2; ++ks) {
                Frag kf;
                kf.q[0] = *(const u32x4*)(ldsK + kr * 72 + ks * 32 + 16 * h16);
                kf.q[1] = *(const u32x4*)(ldsK + kr * 72 + ks * 32 + 16 * h16 + 8);
                a = wmma_bf16(kf, qf[ks], a);
            }
            sc[t] = a;
        }

        // ---- online softmax (rows = q = lane axis)
        float mx = -INFINITY;
#pragma unroll
        for (int t = 0; t < 8; ++t)
#pragma unroll
            for (int j = 0; j < 8; ++j) {
                float v = sc[t][j] * 0.125f + ldsBias[t * 16 + j + 8 * h16];
                sc[t][j] = v;
                mx = fmaxf(mx, v);
            }
        mx = fmaxf(mx, __shfl_xor(mx, 16));
        const float newm = fmaxf(mrun, mx);
        const float corr = __expf(mrun - newm);
        float psum = 0.0f;
#pragma unroll
        for (int t = 0; t < 8; ++t)
#pragma unroll
            for (int j = 0; j < 8; ++j) {
                float pv = __expf(sc[t][j] - newm);
                sc[t][j] = pv;
                psum += pv;
            }
        psum += __shfl_xor(psum, 16);
        lrun = lrun * corr + psum;
        mrun = newm;
#pragma unroll
        for (int dt = 0; dt < 4; ++dt)
#pragma unroll
            for (int e = 0; e < 8; ++e) ctx[dt][e] *= corr;

        // ---- P^T as B operand: half-swap shuffle + bf16 pack
        Frag pf[4];
#pragma unroll
        for (int ks = 0; ks < 4; ++ks) {
#pragma unroll
            for (int j = 0; j < 8; ++j) {
                const int jj = (2 * j) & 7;
                float a0 = sc[2 * ks][jj],     b0 = sc[2 * ks + 1][jj];
                float a1 = sc[2 * ks][jj + 1], b1 = sc[2 * ks + 1][jj + 1];
                float recv0 = __shfl_xor(h16 ? a0 : b0, 16);
                float recv1 = __shfl_xor(h16 ? a1 : b1, 16);
                float own0 = h16 ? b0 : a0;
                float own1 = h16 ? b1 : a1;
                const bool cond = (j < 4) ? (h16 == 0) : (h16 == 1);
                pf[ks].u[j] = pk2bf(cond ? own0 : recv0, cond ? own1 : recv1);
            }
        }

        // ---- ctx^T += V^T * P^T : 4 d-tiles x 4 k-steps
#pragma unroll
        for (int dt = 0; dt < 4; ++dt) {
            const int dr = dt * 16 + l16;
#pragma unroll
            for (int ks = 0; ks < 4; ++ks) {
                Frag vf;
                vf.q[0] = *(const u32x4*)(ldsVT + dr * 136 + ks * 32 + 16 * h16);
                vf.q[1] = *(const u32x4*)(ldsVT + dr * 136 + ks * 32 + 16 * h16 + 8);
                ctx[dt] = wmma_bf16(vf, pf[ks], ctx[dt]);
            }
        }
    }

    // ---- normalize, transpose via LDS (reuse buffer 0), coalesced store
    float* ldsOut = (float*)smem;                 // [128][65]
    const float inv = 1.0f / lrun;
    __syncthreads();
#pragma unroll
    for (int dt = 0; dt < 4; ++dt)
#pragma unroll
        for (int j = 0; j < 8; ++j) {
            const int d = dt * 16 + j + 8 * h16;
            ldsOut[(wv * 16 + l16) * 65 + d] = ctx[dt][j] * inv;
        }
    __syncthreads();
    {
        const int r  = tid >> 1;
        const int db = (tid & 1) * 32;
        float* dst = out + ((size_t)b * SEQ + qblk * 128 + r) * HDIM + h * HD + db;
        const float* src = ldsOut + r * 65 + db;
#pragma unroll
        for (int i = 0; i < 8; ++i) {
            float4 f;
            f.x = src[4 * i + 0]; f.y = src[4 * i + 1];
            f.z = src[4 * i + 2]; f.w = src[4 * i + 3];
            ((float4*)dst)[i] = f;
        }
    }
}

// ---------------------------------------------------------------------------
extern "C" void kernel_launch(void* const* d_in, const int* in_sizes, int n_in,
                              void* d_out, int out_size, void* d_ws, size_t ws_size,
                              hipStream_t stream) {
    (void)in_sizes; (void)n_in; (void)out_size; (void)ws_size;
    const float*     X    = (const float*)d_in[0];
    const long long* mask = (const long long*)d_in[1];
    const float* Wq = (const float*)d_in[2];
    const float* bq = (const float*)d_in[3];
    const float* Wk = (const float*)d_in[4];
    const float* bk = (const float*)d_in[5];
    const float* Wv = (const float*)d_in[6];
    const float* bv = (const float*)d_in[7];
    float* out = (float*)d_out;

    unsigned short* qws = (unsigned short*)d_ws;            // 16 MB
    unsigned short* kws = qws + (size_t)NB * SEQ * HDIM;    // 16 MB
    unsigned short* vws = kws + (size_t)NB * SEQ * HDIM;    // 16 MB (transposed)

    dim3 blk(256);
    dim3 g1(64, 8, 3);
    mhsa_qkv_kernel<<<g1, blk, 0, stream>>>(X, Wq, bq, Wk, bk, Wv, bv, qws, kws, vws);
    dim3 g2(16, 16, 4);
    mhsa_attn_kernel<<<g2, blk, 0, stream>>>(qws, kws, vws, mask, out);
}
